// InstrumentedMultiheadAttention_71992241815955
// MI455X (gfx1250) — matrix operand order repaired and checked by reading, not compiled, not run on hardware
//
#include <hip/hip_runtime.h>

// Problem constants (from reference)
#define Bq  4
#define Tq  1024
#define Cq  1024
#define Hq  16
#define Dq  64
#define N3C 3072   // 3*C

typedef __bf16 bf16_t;
typedef bf16_t v16bf __attribute__((ext_vector_type(16)));
typedef bf16_t v8bf  __attribute__((ext_vector_type(8)));
typedef float  v8f   __attribute__((ext_vector_type(8)));

// ---------------------------------------------------------------------------
// WMMA fragment helpers (CDNA5 16x16x32 bf16, wave32).
// A (16x32, MxK): per-lane 16 bf16 = two contiguous 8-elt runs at
//   k0 + 8*(lane>>4) and +16; row M = lane&15.
// B (32x16, KxN): lane n = lane&15 holds column n; two contiguous 8-elt runs
//   at k0 + 8*(lane>>4) and +16. Source must be [N][K] row-major.
// C/D (16x16 f32): vgpr r on lane l holds (M = r + 8*(l>>4), N = l&15).
// ---------------------------------------------------------------------------
__device__ __forceinline__ v16bf load_fragA(const bf16_t* base, int ld, int k0, int lane) {
  const bf16_t* p = base + (size_t)(lane & 15) * ld + k0 + 8 * (lane >> 4);
  union { v16bf v; v8bf h[2]; } u;
  u.h[0] = *(const v8bf*)(p);
  u.h[1] = *(const v8bf*)(p + 16);
  return u.v;
}

__device__ __forceinline__ v16bf load_fragB(const bf16_t* base, int ld, int k0, int lane) {
  const bf16_t* p = base + (size_t)(lane & 15) * ld + k0 + 8 * (lane >> 4);
  union { v16bf v; v8bf h[2]; } u;
  u.h[0] = *(const v8bf*)(p);
  u.h[1] = *(const v8bf*)(p + 16);
  return u.v;
}

__device__ __forceinline__ v8f wmma_bf16(v16bf a, v16bf b, v8f c) {
  return __builtin_amdgcn_wmma_f32_16x16x32_bf16(false, a, false, b, (short)0, c, false, false);
}

// Async global->LDS copy of 16 bytes per lane (CDNA5 GV addressing).
// VDST = VGPR with LDS byte address, VADDR = 64-bit global address.
__device__ __forceinline__ void async_copy_b128(unsigned lds_addr, const void* gptr) {
  asm volatile("global_load_async_to_lds_b128 %0, %1, off"
               :: "v"(lds_addr), "v"((unsigned long long)(uintptr_t)gptr)
               : "memory");
}
__device__ __forceinline__ void async_wait0() {
  asm volatile("s_wait_asynccnt 0" ::: "memory");
}

// ---------------------------------------------------------------------------
// fp32 -> bf16 conversion (grid-stride)
// ---------------------------------------------------------------------------
__global__ void cvt_f32_to_bf16(const float* __restrict__ in, bf16_t* __restrict__ out, int n) {
  for (int i = blockIdx.x * blockDim.x + threadIdx.x; i < n; i += gridDim.x * blockDim.x)
    out[i] = (bf16_t)in[i];
}

// ---------------------------------------------------------------------------
// QKV GEMM: [B*T, C] x [3C, C]^T + bias -> scatter bf16 q/k/v in [B,H,T,D]
// 64x64 tile per wave: 4 held A frags x 4 streamed B frags -> 16 WMMAs per
// 8 fragment loads (4x arithmetic intensity of a 16x16 tile).
// ---------------------------------------------------------------------------
__global__ __launch_bounds__(256) void qkv_gemm(const bf16_t* __restrict__ xb,
                                                const bf16_t* __restrict__ wb,
                                                const float* __restrict__ bias,
                                                bf16_t* __restrict__ qr,
                                                bf16_t* __restrict__ kr,
                                                bf16_t* __restrict__ vr) {
  const int lane = threadIdx.x & 31;
  const int wtile = blockIdx.x * 8 + (threadIdx.x >> 5);   // 0..3071
  const int mt = wtile / (N3C / 64);
  const int nt = wtile % (N3C / 64);
  const int m0 = mt * 64, n0 = nt * 64;

  const bf16_t* atile = xb + (size_t)m0 * Cq;
  const bf16_t* btile = wb + (size_t)n0 * Cq;

  v8f acc[16];
  for (int t = 0; t < 16; ++t) acc[t] = (v8f){};

#pragma unroll 1
  for (int k0 = 0; k0 < Cq; k0 += 32) {
    if (k0 + 64 < Cq) {
      __builtin_prefetch(atile + (size_t)(lane & 15) * Cq + k0 + 64, 0, 0);
      __builtin_prefetch(btile + (size_t)(lane & 15) * Cq + k0 + 64, 0, 0);
    }
    v16bf a[4];
#pragma unroll
    for (int i = 0; i < 4; ++i)
      a[i] = load_fragA(atile + (size_t)i * 16 * Cq, Cq, k0, lane);
#pragma unroll
    for (int j = 0; j < 4; ++j) {
      v16bf b = load_fragB(btile + (size_t)j * 16 * Cq, Cq, k0, lane);
#pragma unroll
      for (int i = 0; i < 4; ++i)
        acc[i * 4 + j] = wmma_bf16(a[i], b, acc[i * 4 + j]);
    }
  }

  const int hi = lane >> 4;
#pragma unroll
  for (int j = 0; j < 4; ++j) {
    const int n = n0 + j * 16 + (lane & 15);
    const int which = n >> 10;          // 0=q,1=k,2=v
    const int c = n & 1023;
    const int h = c >> 6, d = c & 63;
    const float bn = bias[n];
    bf16_t* dst = (which == 0) ? qr : ((which == 1) ? kr : vr);
#pragma unroll
    for (int i = 0; i < 4; ++i)
      for (int r = 0; r < 8; ++r) {
        const int m = m0 + i * 16 + r + 8 * hi;
        const int bi = m >> 10, t = m & 1023;
        dst[(((size_t)bi * Hq + h) * Tq + t) * Dq + d] = (bf16_t)(acc[i * 4 + j][r] + bn);
      }
  }
}

// ---------------------------------------------------------------------------
// RoPE in-place on q,k (bf16, fp32 math). One thread per (b,h,t,d<32) pair.
// ---------------------------------------------------------------------------
__global__ void rope_kernel(bf16_t* __restrict__ q, bf16_t* __restrict__ k,
                            const float* __restrict__ sn, const float* __restrict__ cs) {
  const int i = blockIdx.x * blockDim.x + threadIdx.x;
  if (i >= Bq * Hq * Tq * 32) return;
  const int d = i & 31;
  const int t = (i >> 5) & 1023;
  const int bh = i >> 15;
  const size_t off = ((size_t)bh * Tq + t) * Dq;
  const float c1 = cs[t * Dq + d],      s1 = sn[t * Dq + d];
  const float c2 = cs[t * Dq + d + 32], s2 = sn[t * Dq + d + 32];
  {
    float u1 = (float)q[off + d], u2 = (float)q[off + d + 32];
    q[off + d]      = (bf16_t)(u1 * c1 - u2 * s1);
    q[off + d + 32] = (bf16_t)(u2 * c2 + u1 * s2);
  }
  {
    float u1 = (float)k[off + d], u2 = (float)k[off + d + 32];
    k[off + d]      = (bf16_t)(u1 * c1 - u2 * s1);
    k[off + d + 32] = (bf16_t)(u2 * c2 + u1 * s2);
  }
}

// v [B,H,T,D] -> vT [B,H,D,T] (so attn@v B-operand is [N][K] row-major)
__global__ void transpose_v(const bf16_t* __restrict__ v, bf16_t* __restrict__ vT) {
  const int i = blockIdx.x * blockDim.x + threadIdx.x;
  if (i >= Bq * Hq * Tq * Dq) return;
  const int d = i & 63;
  const int t = (i >> 6) & 1023;
  const int bh = i >> 16;
  vT[((size_t)bh * Dq + d) * Tq + t] = v[i];
}

// ---------------------------------------------------------------------------
// Row-max pass: block = (b, h, 128 q-rows), 8 waves (one 16-row tile each).
// K chunks of 128 rows are staged cooperatively into LDS with async
// global->LDS b128 copies (shared across the 8 waves), then each wave runs
// 2 WMMAs per 16-col s-tile, fuses scale + rel_bias, and tracks the running
// row max (wave32 shfl_xor reduction). Nothing else is written -> the raw
// score tensor never touches HBM.
// ---------------------------------------------------------------------------
__global__ __launch_bounds__(256) void scores_rowmax(const bf16_t* __restrict__ qb,
                                                     const bf16_t* __restrict__ kb,
                                                     const float* __restrict__ rel_bias,
                                                     float* __restrict__ rowmax) {
  __shared__ __align__(16) bf16_t k_lds[128 * Dq];   // 16 KB chunk of K
  const int tid = threadIdx.x;
  const int lane = tid & 31;
  const int wv = tid >> 5;                 // 0..7
  const int blk = blockIdx.x;              // 512 = B*H*(T/128)
  const int tqb = blk & 7;
  const int h = (blk >> 3) & 15;
  const int b = blk >> 7;
  const int tq = tqb * 8 + wv;             // 16-row tile index 0..63
  const int hi = lane >> 4;

  const bf16_t* qrow = qb + (((size_t)b * Hq + h) * Tq + tq * 16) * Dq;
  const bf16_t* kbase = kb + ((size_t)b * Hq + h) * Tq * Dq;
  const float* brow = rel_bias + ((size_t)h * Tq + tq * 16) * Tq;

  const v16bf a0 = load_fragA(qrow, Dq, 0, lane);
  const v16bf a1 = load_fragA(qrow, Dq, 32, lane);
  const float scale = 0.125f;              // D^-0.5

  float mx[8];
  for (int r = 0; r < 8; ++r) mx[r] = -3.0e38f;

  const unsigned lds_base = (unsigned)(uintptr_t)(&k_lds[0]);

#pragma unroll 1
  for (int kc = 0; kc < 8; ++kc) {
    const char* src = (const char*)(kbase + (size_t)kc * 128 * Dq);
    // 16 KB chunk: 256 threads x 4 x 16B async copies
#pragma unroll
    for (int j = 0; j < 4; ++j) {
      const unsigned off = (unsigned)tid * 16u + (unsigned)j * 4096u;
      async_copy_b128(lds_base + off, src + off);
    }
    async_wait0();
    __syncthreads();

#pragma unroll 1
    for (int st = 0; st < 8; ++st) {
      const bf16_t* kt = k_lds + st * 16 * Dq;
      v8f acc = {};
      acc = wmma_bf16(a0, load_fragB(kt, Dq, 0, lane), acc);
      acc = wmma_bf16(a1, load_fragB(kt, Dq, 32, lane), acc);
      const int n = kc * 128 + st * 16 + (lane & 15);
      for (int r = 0; r < 8; ++r) {
        const int row = r + 8 * hi;
        const float v = acc[r] * scale + brow[(size_t)row * Tq + n];
        mx[r] = fmaxf(mx[r], v);
      }
    }
    __syncthreads();
  }

  for (int off = 1; off < 16; off <<= 1)
    for (int r = 0; r < 8; ++r)
      mx[r] = fmaxf(mx[r], __shfl_xor(mx[r], off, 32));
  for (int r = 0; r < 8; ++r)
    rowmax[((size_t)b * Hq + h) * Tq + tq * 16 + r + 8 * hi] = mx[r];
}

// ---------------------------------------------------------------------------
// Softmax + ctx: one wave per (b,h,16-row block). Recomputes scores with
// WMMA (cheap vs. spilling 268 MB of raw scores), exponentiates into bf16
// LDS (WMMA-A layout loadable), writes normalized attn fp32 exactly once,
// then ctx = (e @ vT^T) * gain / sum via 128 WMMAs from LDS A-fragments.
// ---------------------------------------------------------------------------
#define ELD (Tq + 8)   // padded LDS row stride (elements)

__global__ __launch_bounds__(32) void softmax_ctx(const bf16_t* __restrict__ qb,
                                                  const bf16_t* __restrict__ kb,
                                                  const float* __restrict__ rel_bias,
                                                  const float* __restrict__ rowmax,
                                                  const bf16_t* __restrict__ vT,
                                                  const float* __restrict__ gains,
                                                  float* __restrict__ attn,
                                                  float* __restrict__ ctx,
                                                  bf16_t* __restrict__ merged) {
  __shared__ __align__(16) bf16_t e_lds[16 * ELD];
  __shared__ float inv_sum[16];

  const int lane = threadIdx.x;
  const int wid = blockIdx.x;
  const int tq = wid & 63;
  const int h  = (wid >> 6) & 15;
  const int b  = wid >> 10;
  const int hi = lane >> 4;

  const bf16_t* qrow = qb + (((size_t)b * Hq + h) * Tq + tq * 16) * Dq;
  const bf16_t* kbase = kb + ((size_t)b * Hq + h) * Tq * Dq;
  const float* brow = rel_bias + ((size_t)h * Tq + tq * 16) * Tq;
  const float* rmx = rowmax + ((size_t)b * Hq + h) * Tq + tq * 16;
  float* srow = attn + (((size_t)b * Hq + h) * Tq + tq * 16) * Tq;

  const v16bf a0 = load_fragA(qrow, Dq, 0, lane);
  const v16bf a1 = load_fragA(qrow, Dq, 32, lane);

  float p[8];
  for (int r = 0; r < 8; ++r) p[r] = 0.f;

  // recompute scores, exp into LDS, accumulate row sums
#pragma unroll 1
  for (int st = 0; st < Tq / 16; ++st) {
    const bf16_t* kt = kbase + (size_t)st * 16 * Dq;
    v8f acc = {};
    acc = wmma_bf16(a0, load_fragB(kt, Dq, 0, lane), acc);
    acc = wmma_bf16(a1, load_fragB(kt, Dq, 32, lane), acc);
    const int n = st * 16 + (lane & 15);
    for (int r = 0; r < 8; ++r) {
      const int row = r + 8 * hi;
      const float v = acc[r] * 0.125f + brow[(size_t)row * Tq + n];
      const float e = __expf(v - rmx[row]);
      e_lds[row * ELD + n] = (bf16_t)e;
      p[r] += e;
    }
  }
  for (int off = 1; off < 16; off <<= 1)
    for (int r = 0; r < 8; ++r)
      p[r] += __shfl_xor(p[r], off, 32);
  for (int r = 0; r < 8; ++r)
    inv_sum[r + 8 * hi] = 1.0f / p[r];
  __syncthreads();

  // normalized attn (fp32), single HBM write
  for (int row = 0; row < 16; ++row) {
    const float is = inv_sum[row];
    for (int c0 = lane; c0 < Tq; c0 += 32)
      srow[(size_t)row * Tq + c0] = (float)e_lds[row * ELD + c0] * is;
  }
  __syncthreads();

  // ctx GEMM: 4 d-tiles x 32 K-steps of WMMA, A fragments from LDS
  const float g = gains[h];
  float* crow = ctx + (((size_t)b * Hq + h) * Tq + tq * 16) * Dq;
  bf16_t* mrow = merged + ((size_t)b * Tq + tq * 16) * Cq + h * Dq;

#pragma unroll 1
  for (int dt = 0; dt < 4; ++dt) {
    const bf16_t* vtile = vT + (((size_t)b * Hq + h) * Dq + dt * 16) * Tq;
    v8f acc = {};
#pragma unroll 1
    for (int k0 = 0; k0 < Tq; k0 += 32) {
      v16bf a = load_fragA(e_lds, ELD, k0, lane);
      v16bf bfr = load_fragB(vtile, Tq, k0, lane);
      acc = wmma_bf16(a, bfr, acc);
    }
    const int d = dt * 16 + (lane & 15);
    for (int r = 0; r < 8; ++r) {
      const int row = r + 8 * hi;
      const float val = acc[r] * inv_sum[row] * g;
      crow[(size_t)row * Dq + d] = val;
      mrow[(size_t)row * Cq + d] = (bf16_t)val;
    }
  }
}

// ---------------------------------------------------------------------------
// Output projection: [B*T, C] x [C, C]^T + bias, 64x64 tile per wave
// ---------------------------------------------------------------------------
__global__ __launch_bounds__(256) void out_gemm(const bf16_t* __restrict__ mb,
                                                const bf16_t* __restrict__ wb,
                                                const float* __restrict__ bias,
                                                float* __restrict__ out) {
  const int lane = threadIdx.x & 31;
  const int wtile = blockIdx.x * 8 + (threadIdx.x >> 5);   // 0..1023
  const int mt = wtile >> 4;        // / (1024/64)
  const int nt = wtile & 15;
  const int m0 = mt * 64, n0 = nt * 64;

  const bf16_t* atile = mb + (size_t)m0 * Cq;
  const bf16_t* btile = wb + (size_t)n0 * Cq;

  v8f acc[16];
  for (int t = 0; t < 16; ++t) acc[t] = (v8f){};

#pragma unroll 1
  for (int k0 = 0; k0 < Cq; k0 += 32) {
    if (k0 + 64 < Cq) {
      __builtin_prefetch(atile + (size_t)(lane & 15) * Cq + k0 + 64, 0, 0);
      __builtin_prefetch(btile + (size_t)(lane & 15) * Cq + k0 + 64, 0, 0);
    }
    v16bf a[4];
#pragma unroll
    for (int i = 0; i < 4; ++i)
      a[i] = load_fragA(atile + (size_t)i * 16 * Cq, Cq, k0, lane);
#pragma unroll
    for (int j = 0; j < 4; ++j) {
      v16bf b = load_fragB(btile + (size_t)j * 16 * Cq, Cq, k0, lane);
#pragma unroll
      for (int i = 0; i < 4; ++i)
        acc[i * 4 + j] = wmma_bf16(a[i], b, acc[i * 4 + j]);
    }
  }

  const int hi = lane >> 4;
#pragma unroll
  for (int j = 0; j < 4; ++j) {
    const int n = n0 + j * 16 + (lane & 15);
    const float bn = bias[n];
#pragma unroll
    for (int i = 0; i < 4; ++i)
      for (int r = 0; r < 8; ++r)
        out[(size_t)(m0 + i * 16 + r + 8 * hi) * Cq + n] = acc[i * 4 + j][r] + bn;
  }
}

// ---------------------------------------------------------------------------
// Launch
// ---------------------------------------------------------------------------
extern "C" void kernel_launch(void* const* d_in, const int* in_sizes, int n_in,
                              void* d_out, int out_size, void* d_ws, size_t ws_size,
                              hipStream_t stream) {
  const float* x       = (const float*)d_in[0];
  const float* qkv_w   = (const float*)d_in[1];
  const float* qkv_b   = (const float*)d_in[2];
  const float* out_w   = (const float*)d_in[3];
  const float* out_b   = (const float*)d_in[4];
  const float* rel_b   = (const float*)d_in[5];
  const float* sn      = (const float*)d_in[6];
  const float* cs      = (const float*)d_in[7];
  const float* gains   = (const float*)d_in[8];

  float* out_p  = (float*)d_out;                              // [B,T,C]
  float* ctx_p  = out_p + (size_t)Bq * Tq * Cq;               // [B,H,T,D]
  float* attn_p = ctx_p + (size_t)Bq * Hq * Tq * Dq;          // [B,H,T,T]

  char* ws = (char*)d_ws;
  bf16_t* xb     = (bf16_t*)(ws);                             // 8 MB
  bf16_t* wqkv   = (bf16_t*)(ws + (8u << 20));                // 6 MB
  bf16_t* wout   = (bf16_t*)(ws + (14u << 20));               // 2 MB
  bf16_t* q_bf   = (bf16_t*)(ws + (16u << 20));               // 8 MB
  bf16_t* k_bf   = (bf16_t*)(ws + (24u << 20));               // 8 MB
  bf16_t* v_bf   = (bf16_t*)(ws + (32u << 20));               // 8 MB
  bf16_t* vT_bf  = (bf16_t*)(ws + (40u << 20));               // 8 MB
  bf16_t* merged = (bf16_t*)(ws + (48u << 20));               // 8 MB
  float*  rowmax = (float*) (ws + (56u << 20));               // 256 KB

  // 1) fp32 -> bf16 of x, qkv_w, out_w
  cvt_f32_to_bf16<<<1024, 256, 0, stream>>>(x, xb, Bq * Tq * Cq);
  cvt_f32_to_bf16<<<1024, 256, 0, stream>>>(qkv_w, wqkv, N3C * Cq);
  cvt_f32_to_bf16<<<1024, 256, 0, stream>>>(out_w, wout, Cq * Cq);

  // 2) QKV projection: 64x64 tiles, 8 waves/block -> (64*48)/8 blocks
  qkv_gemm<<<384, 256, 0, stream>>>(xb, wqkv, qkv_b, q_bf, k_bf, v_bf);

  // 3) RoPE + v transpose
  rope_kernel<<<(Bq * Hq * Tq * 32 + 255) / 256, 256, 0, stream>>>(q_bf, k_bf, sn, cs);
  transpose_v<<<(Bq * Hq * Tq * Dq + 255) / 256, 256, 0, stream>>>(v_bf, vT_bf);

  // 4) Row-max with async-LDS staged K chunks (B*H*(T/128) blocks)
  scores_rowmax<<<512, 256, 0, stream>>>(q_bf, k_bf, rel_b, rowmax);

  // 5) Softmax + ctx (score recompute; one wave per block; ~33 KB LDS)
  softmax_ctx<<<4096, 32, 0, stream>>>(q_bf, k_bf, rel_b, rowmax, vT_bf, gains,
                                       attn_p, ctx_p, merged);

  // 6) Output projection: 64x64 tiles -> (64*16)/8 blocks
  out_gemm<<<128, 256, 0, stream>>>(merged, wout, out_b, out_p);
}
